// EdgeVariantGNN_Lin_8967891714102
// MI455X (gfx1250) — matrix operand order repaired and checked by reading, not compiled, not run on hardware
//
#include <hip/hip_runtime.h>
#include <hip/hip_bf16.h>

// ---------- types ----------
typedef __attribute__((ext_vector_type(16))) __bf16 v16bf;
typedef __attribute__((ext_vector_type(8)))  float  v8f;

#define NNODES 512
#define BATCH  64
#define NSQ    (512 * 512)

// ---------- helpers ----------
static __device__ __forceinline__ unsigned short bf16_bits(float f) {
  unsigned u = __builtin_bit_cast(unsigned, f);
  u += 0x7FFFu + ((u >> 16) & 1u);          // round-to-nearest-even
  return (unsigned short)(u >> 16);
}
static __device__ __forceinline__ __bf16 bf16_of(unsigned short u) {
  return __builtin_bit_cast(__bf16, u);
}
static __device__ __forceinline__ __bf16 cvt_mask(float v, unsigned mw, int bit) {
  if (!((mw >> bit) & 1u)) v = 0.0f;
  return bf16_of(bf16_bits(v));
}
static __device__ __forceinline__ __bf16 cvt_plain(float v) {
  return bf16_of(bf16_bits(v));
}

// ---------- build packed sparsity mask: sp = support(S)+I, kill [M:,M:] block ----------
__global__ void mask_kernel(const float* __restrict__ S, unsigned* __restrict__ mask) {
  int w = blockIdx.x * blockDim.x + threadIdx.x;
  if (w >= 8192) return;                    // 512*512/32 words
  int m  = w >> 4;
  int n0 = (w & 15) << 5;
  unsigned bits = 0u;
  for (int i = 0; i < 32; ++i) {
    int n = n0 + i;
    float s = S[m * NNODES + n];
    bool on = (fabsf(s) + ((m == n) ? 1.0f : 0.0f)) > 1e-9f;
    if (m >= 256 && n >= 256) on = false;   // hybrid EV: M = 256
    if (on) bits |= (1u << i);
  }
  mask[w] = bits;
}

// ---------- f32 -> bf16 bulk convert ----------
__global__ void f32_to_bf16_kernel(const float* __restrict__ src,
                                   unsigned short* __restrict__ dst, int n) {
  int i = blockIdx.x * blockDim.x + threadIdx.x;
  if (i < n) dst[i] = bf16_bits(src[i]);
}

// ---------- generic bf16-WMMA GEMM over "chains" ----------
// Per chain c: Y(Mrows x 64) = A_c(Mrows x Kdim, fp32, MASKED => elementwise sp mask) * B_c(Kdim x 64, bf16)
// One wave = one 16-row tile x all 64 batch columns (4 WMMA accumulators).
// A per-lane layout (16-bit A, 16x32): lanes<16 K={0..7,16..23}, lanes>=16 K={8..15,24..31}.
// B per-lane layout (16-bit B, 32x16): lane holds column n=lane%16, K={8h..8h+7,16+8h..16+8h+7}, h=lane/16 (same as A).
template <bool MASKED>
__global__ __launch_bounds__(256) void gemm_bf16_wmma(
    const float* __restrict__ A, long long asf, long long asg, int Ga,
    const unsigned* __restrict__ maskbits,          // used only when MASKED (Kdim==512 pattern)
    const unsigned short* __restrict__ Bact, long long bstride, int Gb,
    float* __restrict__ yacc, int yacc_overwrite,   // per-chain f32 accumulator [c][col][row]
    unsigned short* __restrict__ zout,              // per-chain bf16 out [c][col][row]
    float* __restrict__ fout,                       // per-chain f32 out  [c][col][row]
    const float* __restrict__ bias,                 // bias[row], applied to zout/fout
    int Mrows, int Kdim, int nchains)
{
  const int lane  = threadIdx.x & 31;
  const int wid   = (blockIdx.x * blockDim.x + threadIdx.x) >> 5;
  const int mtiles = Mrows >> 4;
  if (wid >= nchains * mtiles) return;              // whole-wave uniform exit (EXEC stays full)
  const int chain = wid / mtiles;
  const int mtile = wid - chain * mtiles;
  const int m0  = mtile << 4;
  const int rl  = lane & 15;
  const int sel = lane >> 4;
  const int row = m0 + rl;

  const float* Ac = A + (long long)(chain / Ga) * asf + (long long)(chain % Ga) * asg
                      + (long long)row * Kdim;
  const unsigned short* Bc = Bact + (long long)(chain % Gb) * bstride;
  const unsigned* mrow = MASKED ? (maskbits + (row << 4)) : nullptr;
  const int kchunks = Kdim >> 5;

  v8f acc[4] = {};

  for (int kc = 0; kc < kchunks; ++kc) {
    const float* ap = Ac + (kc << 5) + (sel << 3);
    __builtin_prefetch(ap + 32, 0, 1);              // next A chunk -> global_prefetch
    float4 w0 = *(const float4*)ap;
    float4 w1 = *(const float4*)(ap + 4);
    float4 w2 = *(const float4*)(ap + 16);
    float4 w3 = *(const float4*)(ap + 20);
    v16bf a;
    if (MASKED) {
      unsigned mw = mrow[kc] >> (sel << 3);         // one word covers this row's 32 cols
      a[0]  = cvt_mask(w0.x, mw, 0);  a[1]  = cvt_mask(w0.y, mw, 1);
      a[2]  = cvt_mask(w0.z, mw, 2);  a[3]  = cvt_mask(w0.w, mw, 3);
      a[4]  = cvt_mask(w1.x, mw, 4);  a[5]  = cvt_mask(w1.y, mw, 5);
      a[6]  = cvt_mask(w1.z, mw, 6);  a[7]  = cvt_mask(w1.w, mw, 7);
      a[8]  = cvt_mask(w2.x, mw, 16); a[9]  = cvt_mask(w2.y, mw, 17);
      a[10] = cvt_mask(w2.z, mw, 18); a[11] = cvt_mask(w2.w, mw, 19);
      a[12] = cvt_mask(w3.x, mw, 20); a[13] = cvt_mask(w3.y, mw, 21);
      a[14] = cvt_mask(w3.z, mw, 22); a[15] = cvt_mask(w3.w, mw, 23);
    } else {
      a[0]  = cvt_plain(w0.x);  a[1]  = cvt_plain(w0.y);
      a[2]  = cvt_plain(w0.z);  a[3]  = cvt_plain(w0.w);
      a[4]  = cvt_plain(w1.x);  a[5]  = cvt_plain(w1.y);
      a[6]  = cvt_plain(w1.z);  a[7]  = cvt_plain(w1.w);
      a[8]  = cvt_plain(w2.x);  a[9]  = cvt_plain(w2.y);
      a[10] = cvt_plain(w2.z);  a[11] = cvt_plain(w2.w);
      a[12] = cvt_plain(w3.x);  a[13] = cvt_plain(w3.y);
      a[14] = cvt_plain(w3.z);  a[15] = cvt_plain(w3.w);
    }

    const int bkoff = (kc << 5) + (sel << 3);
    #pragma unroll
    for (int bt = 0; bt < 4; ++bt) {
      const unsigned short* bp = Bc + (long long)((bt << 4) + rl) * Kdim + bkoff;
      v16bf bv;                                     // two 16B halves: k 8h..8h+7, 16+8h..16+8h+7
      *(uint4*)&bv = *(const uint4*)bp;
      *((uint4*)&bv + 1) = *(const uint4*)(bp + 16);
      acc[bt] = __builtin_amdgcn_wmma_f32_16x16x32_bf16(
                    false, a, false, bv, (short)0, acc[bt], false, false);
    }
  }

  // D layout: VGPR r -> row m0 + r + sel*8, col = bt*16 + (lane&15)  => 8 contiguous rows/lane
  const int mr = m0 + (sel << 3);
  float bv8[8];
  #pragma unroll
  for (int r = 0; r < 8; ++r) bv8[r] = bias ? bias[mr + r] : 0.0f;

  #pragma unroll
  for (int bt = 0; bt < 4; ++bt) {
    const int col = (bt << 4) + rl;
    const long long off = ((long long)chain * BATCH + col) * Mrows + mr;
    if (yacc) {
      float* yp = yacc + off;
      #pragma unroll
      for (int r = 0; r < 8; ++r) {
        float v = acc[bt][r];
        if (!yacc_overwrite) v += yp[r];
        yp[r] = v;
      }
    }
    if (zout) {
      unsigned p0 = (unsigned)bf16_bits(acc[bt][0] + bv8[0]) | ((unsigned)bf16_bits(acc[bt][1] + bv8[1]) << 16);
      unsigned p1 = (unsigned)bf16_bits(acc[bt][2] + bv8[2]) | ((unsigned)bf16_bits(acc[bt][3] + bv8[3]) << 16);
      unsigned p2 = (unsigned)bf16_bits(acc[bt][4] + bv8[4]) | ((unsigned)bf16_bits(acc[bt][5] + bv8[5]) << 16);
      unsigned p3 = (unsigned)bf16_bits(acc[bt][6] + bv8[6]) | ((unsigned)bf16_bits(acc[bt][7] + bv8[7]) << 16);
      *(uint4*)(zout + off) = make_uint4(p0, p1, p2, p3);
    }
    if (fout) {
      float* fp = fout + off;
      #pragma unroll
      for (int r = 0; r < 8; ++r) fp[r] = acc[bt][r] + bv8[r];
    }
  }
}

// ---------- layer-1 reduction: y1 = yacc1 + sum_k wLSI1[f,k]*S^k x + b1 ----------
__global__ void reduce1_kernel(const float* __restrict__ yacc1, const float* __restrict__ wLSI1,
                               const float* __restrict__ b1,
                               const float* __restrict__ s0, const float* __restrict__ s1,
                               const float* __restrict__ s2,
                               float* __restrict__ y1f, unsigned short* __restrict__ y1bf) {
  int idx = blockIdx.x * blockDim.x + threadIdx.x;
  if (idx >= 8 * BATCH * NNODES) return;
  int f  = idx >> 15;                 // / (64*512)
  int bm = idx & 32767;
  float v = yacc1[idx]
          + wLSI1[f * 3 + 0] * s0[bm]
          + wLSI1[f * 3 + 1] * s1[bm]
          + wLSI1[f * 3 + 2] * s2[bm]
          + b1[f];
  y1f[idx]  = v;
  y1bf[idx] = bf16_bits(v);
}

// ---------- layer-2 reduction: y2 = sum_g yacc2 + sum_{k,g} wLSI2[f,k,g]*S^k y1 + b2 ----------
__global__ void reduce2_kernel(const float* __restrict__ yacc2, const float* __restrict__ wLSI2,
                               const float* __restrict__ b2,
                               const float* __restrict__ s0, const float* __restrict__ s1,
                               const float* __restrict__ s2,
                               unsigned short* __restrict__ y2bf) {
  int idx = blockIdx.x * blockDim.x + threadIdx.x;
  if (idx >= 8 * BATCH * NNODES) return;
  int f  = idx >> 15;
  int bm = idx & 32767;
  int b  = bm >> 9;
  int m  = bm & 511;
  float v = b2[f];
  #pragma unroll
  for (int g = 0; g < 8; ++g) {
    v += yacc2[(((long long)(f * 8 + g)) << 15) + bm];
    int gb = (g << 15) + bm;
    v += wLSI2[(f * 3 + 0) * 8 + g] * s0[gb]
       + wLSI2[(f * 3 + 1) * 8 + g] * s1[gb]
       + wLSI2[(f * 3 + 2) * 8 + g] * s2[gb];
  }
  y2bf[((long long)b) * 4096 + f * 512 + m] = bf16_bits(v);   // [b][f*512+m] for MLP
}

// ---------- host ----------
static inline int gemm_blocks(int nchains, int Mrows) {
  int waves = nchains * (Mrows >> 4);
  return (waves + 7) / 8;                 // 8 waves (256 threads) per block
}

extern "C" void kernel_launch(void* const* d_in, const int* in_sizes, int n_in,
                              void* d_out, int out_size, void* d_ws, size_t ws_size,
                              hipStream_t stream) {
  const float* x     = (const float*)d_in[0];
  const float* S     = (const float*)d_in[1];
  const float* wEV1  = (const float*)d_in[2];
  const float* wLSI1 = (const float*)d_in[3];
  const float* b1    = (const float*)d_in[4];
  const float* wEV2  = (const float*)d_in[5];
  const float* wLSI2 = (const float*)d_in[6];
  const float* b2    = (const float*)d_in[7];
  const float* W1    = (const float*)d_in[8];
  const float* bW1   = (const float*)d_in[9];
  const float* W2    = (const float*)d_in[10];
  const float* bW2   = (const float*)d_in[11];
  float* out = (float*)d_out;
  (void)in_sizes; (void)n_in; (void)out_size; (void)ws_size;

  char* wsb = (char*)d_ws;
  size_t off = 0;
  auto alloc = [&](size_t bytes) -> void* {
    void* p = wsb + off;
    off = (off + bytes + 255) & ~(size_t)255;
    return p;
  };
  const size_t BN = (size_t)BATCH * NNODES;           // 32768
  unsigned*       mask  = (unsigned*)      alloc(8192 * 4);
  unsigned short* xbf   = (unsigned short*)alloc(BN * 2);
  unsigned short* za    = (unsigned short*)alloc(64 * BN * 2);     // ping
  unsigned short* zb    = (unsigned short*)alloc(64 * BN * 2);     // pong
  float*          yacc1 = (float*)         alloc(8  * BN * 4);
  float*          yacc2 = (float*)         alloc(64 * BN * 4);
  float*          s1_1  = (float*)         alloc(BN * 4);
  float*          s2_1  = (float*)         alloc(BN * 4);
  unsigned short* s1b_1 = (unsigned short*)alloc(BN * 2);
  float*          y1f   = (float*)         alloc(8 * BN * 4);      // == s0 for layer 2
  unsigned short* y1bf  = (unsigned short*)alloc(8 * BN * 2);
  float*          s1_2  = (float*)         alloc(8 * BN * 4);
  float*          s2_2  = (float*)         alloc(8 * BN * 4);
  unsigned short* s1b_2 = (unsigned short*)alloc(8 * BN * 2);
  unsigned short* y2bf  = (unsigned short*)alloc((size_t)BATCH * 4096 * 2);
  unsigned short* hbf   = (unsigned short*)alloc((size_t)BATCH * 512 * 2);

  const long long NSQL = (long long)NSQ;

  // 0) mask + input convert
  mask_kernel<<<32, 256, 0, stream>>>(S, mask);
  f32_to_bf16_kernel<<<(int)((BN + 255) / 256), 256, 0, stream>>>(x, xbf, (int)BN);

  // 1) layer-1 EV chain: 8 chains (f), G=1, K taps = 3
  //    A mat (f,k): wEV1 + (f*3+k)*NSQ
  gemm_bf16_wmma<true><<<gemm_blocks(8, 512), 256, 0, stream>>>(
      wEV1 + 0 * NSQ, 3 * NSQL, 0, 1, mask, xbf, 0, 1,
      yacc1, 1, za, nullptr, nullptr, 512, 512, 8);
  gemm_bf16_wmma<true><<<gemm_blocks(8, 512), 256, 0, stream>>>(
      wEV1 + 1 * NSQ, 3 * NSQL, 0, 1, mask, za, (long long)BN, 8,
      yacc1, 0, zb, nullptr, nullptr, 512, 512, 8);
  gemm_bf16_wmma<true><<<gemm_blocks(8, 512), 256, 0, stream>>>(
      wEV1 + 2 * NSQ, 3 * NSQL, 0, 1, mask, zb, (long long)BN, 8,
      yacc1, 0, nullptr, nullptr, nullptr, 512, 512, 8);

  // 2) layer-1 LSI: s1 = S x, s2 = S s1 (unmasked S)
  gemm_bf16_wmma<false><<<gemm_blocks(1, 512), 256, 0, stream>>>(
      S, 0, 0, 1, nullptr, xbf, 0, 1,
      nullptr, 0, s1b_1, s1_1, nullptr, 512, 512, 1);
  gemm_bf16_wmma<false><<<gemm_blocks(1, 512), 256, 0, stream>>>(
      S, 0, 0, 1, nullptr, s1b_1, 0, 1,
      nullptr, 0, nullptr, s2_1, nullptr, 512, 512, 1);

  // 3) y1 = yacc1 + LSI + b1   (also bf16 copy for layer-2 inputs)
  reduce1_kernel<<<(8 * (int)BN) / 256, 256, 0, stream>>>(
      yacc1, wLSI1, b1, x, s1_1, s2_1, y1f, y1bf);

  // 4) layer-2 LSI: s1 = S y1 (per g), s2 = S s1
  gemm_bf16_wmma<false><<<gemm_blocks(8, 512), 256, 0, stream>>>(
      S, 0, 0, 1, nullptr, y1bf, (long long)BN, 8,
      nullptr, 0, s1b_2, s1_2, nullptr, 512, 512, 8);
  gemm_bf16_wmma<false><<<gemm_blocks(8, 512), 256, 0, stream>>>(
      S, 0, 0, 1, nullptr, s1b_2, (long long)BN, 8,
      nullptr, 0, nullptr, s2_2, nullptr, 512, 512, 8);

  // 5) layer-2 EV chain: 64 chains c=f*8+g; A mat = wEV2 + (f*24 + k*8 + g)*NSQ
  gemm_bf16_wmma<true><<<gemm_blocks(64, 512), 256, 0, stream>>>(
      wEV2 + (long long)0 * 8 * NSQ, 24 * NSQL, NSQL, 8, mask, y1bf, (long long)BN, 8,
      yacc2, 1, za, nullptr, nullptr, 512, 512, 64);
  gemm_bf16_wmma<true><<<gemm_blocks(64, 512), 256, 0, stream>>>(
      wEV2 + (long long)1 * 8 * NSQ, 24 * NSQL, NSQL, 8, mask, za, (long long)BN, 64,
      yacc2, 0, zb, nullptr, nullptr, 512, 512, 64);
  gemm_bf16_wmma<true><<<gemm_blocks(64, 512), 256, 0, stream>>>(
      wEV2 + (long long)2 * 8 * NSQ, 24 * NSQL, NSQL, 8, mask, zb, (long long)BN, 64,
      yacc2, 0, nullptr, nullptr, nullptr, 512, 512, 64);

  // 6) y2 = sum_g yacc2 + LSI + b2 -> bf16 [b][f*512+m]
  reduce2_kernel<<<(8 * (int)BN) / 256, 256, 0, stream>>>(
      yacc2, wLSI2, b2, y1f, s1_2, s2_2, y2bf);

  // 7) MLP: h = y2 @ W1^T + bW1 ; out = h @ W2^T + bW2
  gemm_bf16_wmma<false><<<gemm_blocks(1, 512), 256, 0, stream>>>(
      W1, 0, 0, 1, nullptr, y2bf, 0, 1,
      nullptr, 0, hbf, nullptr, bW1, 512, 4096, 1);
  gemm_bf16_wmma<false><<<gemm_blocks(1, 128), 256, 0, stream>>>(
      W2, 0, 0, 1, nullptr, hbf, 0, 1,
      nullptr, 0, nullptr, out, bW2, 128, 512, 1);
}
